// AttnBlock3DFix_20143396618755
// MI455X (gfx1250) — matrix operand order repaired and checked by reading, not compiled, not run on hardware
//
#include <hip/hip_runtime.h>
#include <hip/hip_bf16.h>
#include <math.h>

// ---------------------------------------------------------------------------
// AttnBlock3D (GroupNorm -> QKV 1x1 conv -> per-frame softmax attention ->
// output projection + residual) for MI455X / gfx1250, wave32 + WMMA bf16.
// Round 4: same as round 3 but with the 6-arg tensor_load_to_lds builtin
// (clang-23 signature: g0 u32x4, g1 i32x8, i32x4, i32x4, i32x8, i32 cpol).
// ---------------------------------------------------------------------------

#define C_CH 512
#define N_TOK 16384       // t*h*w = 4*64*64
#define HW 4096
#define T_FRAMES 4
#define NUM_GROUPS 32
#define CPG 16            // channels per group
#define GN_EPS 1e-6f

#define LDS_PITCH 40      // elements per tile row in LDS: 32 data + 4-dword pad

typedef __bf16 bf16_t;
typedef __attribute__((ext_vector_type(16))) __bf16 v16bf;
typedef __attribute__((ext_vector_type(8)))  __bf16 v8bf;
typedef __attribute__((ext_vector_type(8)))  float  v8f;
typedef __attribute__((ext_vector_type(4)))  unsigned int u32x4;
typedef __attribute__((ext_vector_type(8)))  int          i32x8;
typedef __attribute__((ext_vector_type(4)))  int          i32x4;

// ---- WMMA helpers ---------------------------------------------------------

static __device__ __forceinline__ v8f wmma_bf16(v16bf a, v16bf b, v8f c) {
  return __builtin_amdgcn_wmma_f32_16x16x32_bf16(
      /*neg_a=*/false, a, /*neg_b=*/false, b,
      /*c_mod=*/(short)0, c, /*reuse_a=*/false, /*reuse_b=*/false);
}

// A-matrix 16x32 bf16 fragment from row-major global storage (uniform base +
// 32-bit element offset). Lanes<16 hold K 0..7 & 16..23, lanes>=16 hold
// K 8..15 & 24..31 -> two contiguous 16B loads.
static __device__ __forceinline__ v16bf load_a_frag(const bf16_t* __restrict__ base,
                                                    unsigned eoff, int lane) {
  unsigned o = eoff + ((lane & 16) ? 8u : 0u);
  v8bf lo = *(const v8bf*)(base + o);
  v8bf hi = *(const v8bf*)(base + o + 16);
  return __builtin_shufflevector(lo, hi, 0, 1, 2, 3, 4, 5, 6, 7,
                                 8, 9, 10, 11, 12, 13, 14, 15);
}

// Same pattern for an LDS row (16B-aligned halves).
static __device__ __forceinline__ v16bf load_a_frag_lds(const bf16_t* p, int lane) {
  const int off = (lane & 16) ? 8 : 0;
  v8bf lo = *(const v8bf*)(p + off);
  v8bf hi = *(const v8bf*)(p + off + 16);
  return __builtin_shufflevector(lo, hi, 0, 1, 2, 3, 4, 5, 6, 7,
                                 8, 9, 10, 11, 12, 13, 14, 15);
}

// B-matrix 32x16 bf16 fragment at this lane's column. Lanes<16 hold
// K 0..7 & 16..23, lanes>=16 hold K 8..15 & 24..31 -> two 16B loads.
static __device__ __forceinline__ v16bf load_b_frag(const bf16_t* __restrict__ base,
                                                    unsigned eoff, int lane) {
  unsigned o = eoff + ((lane & 16) ? 8u : 0u);
  v8bf lo = *(const v8bf*)(base + o);
  v8bf hi = *(const v8bf*)(base + o + 16);
  return __builtin_shufflevector(lo, hi, 0, 1, 2, 3, 4, 5, 6, 7,
                                 8, 9, 10, 11, 12, 13, 14, 15);
}

// LDS variant as two 16B loads (rows are 16B- but not 32B-aligned).
static __device__ __forceinline__ v16bf load_b_frag_lds(const bf16_t* p, int lane) {
  const int off = (lane & 16) ? 8 : 0;
  v8bf lo = *(const v8bf*)(p + off);
  v8bf hi = *(const v8bf*)(p + off + 16);
  return __builtin_shufflevector(lo, hi, 0, 1, 2, 3, 4, 5, 6, 7,
                                 8, 9, 10, 11, 12, 13, 14, 15);
}

// ---- Tensor Data Mover: async 2D bf16 tile load global -> LDS -------------
// Builds the D# per CDNA5 ISA ch.8 and issues TENSOR_LOAD_TO_LDS.
// pad_enable with pad_interval=16dw / pad_amount=4dw turns each 64B tile row
// into an 80B LDS pitch (LDS_PITCH elements): conflict-free fragment reads.

static __device__ __forceinline__ void tdm_load_2d(
    unsigned lds_byte_addr,     // LDS byte offset of tile destination
    const bf16_t* gptr,         // global address of tile origin
    unsigned tensor_d0,         // tensor row length (elements)
    unsigned tensor_d1,         // tensor rows
    unsigned tile_d0,           // tile width (elements)  [32]
    unsigned tile_d1,           // tile height (rows)     [128]
    unsigned d0_stride) {       // tensor row stride (elements)
  unsigned long long ga = (unsigned long long)gptr;
  u32x4 g0;
  g0.x = 0x1u;                                        // count=1 valid user D#
  g0.y = lds_byte_addr;                               // lds_addr [63:32]
  g0.z = (unsigned)(ga & 0xffffffffu);                // global_addr lo
  g0.w = (unsigned)((ga >> 32) & 0x01ffffffu)         // global_addr [56:32]
       | (2u << 30);                                  // type = 2 (image)
  i32x8 g1;
  g1[0] = (int)((1u << 16)        // data_size = 1 -> 2 bytes
              | (1u << 20)        // pad_enable
              | (3u << 22)        // pad_interval: 16 dwords (one 64B row)
              | (3u << 25));      // pad_amount: 4 dwords (16B) -> 80B pitch
  g1[1] = (int)((tensor_d0 & 0xffffu) << 16);                     // dim0 lo
  g1[2] = (int)(((tensor_d0 >> 16) & 0xffffu) |
                ((tensor_d1 & 0xffffu) << 16));                   // dim0 hi | dim1 lo
  g1[3] = (int)(((tensor_d1 >> 16) & 0xffffu) |
                ((tile_d0 & 0xffffu) << 16));                     // dim1 hi | tile0
  g1[4] = (int)(tile_d1 & 0xffffu);                               // tile1 | tile2=0
  g1[5] = (int)d0_stride;                                         // dim0_stride lo
  g1[6] = 0;                                                      // stride hi, d1stride
  g1[7] = 0;
  i32x4 z4 = {0, 0, 0, 0};
  i32x8 z8 = {0, 0, 0, 0, 0, 0, 0, 0};
  __builtin_amdgcn_tensor_load_to_lds(g0, g1, z4, z4, z8, 0);
}

// ---- K0: fp32 -> bf16 weight conversion -----------------------------------

__global__ void cvt_bf16_kernel(const float* __restrict__ src,
                                bf16_t* __restrict__ dst, int n) {
  int i = blockIdx.x * blockDim.x + threadIdx.x;
  if (i < n) dst[i] = (bf16_t)src[i];
}

// ---- K1: GroupNorm statistics (one workgroup per group) -------------------

__global__ void gn_stats_kernel(const float* __restrict__ x,
                                float* __restrict__ stats) {
  const int g = blockIdx.x;
  const float* p = x + (size_t)g * (CPG * N_TOK);
  const int nelem4 = (CPG * N_TOK) / 4;
  float s = 0.f, s2 = 0.f;
  for (int i = threadIdx.x; i < nelem4; i += blockDim.x) {
    float4 v = ((const float4*)p)[i];
    s  += v.x + v.y + v.z + v.w;
    s2 += v.x * v.x + v.y * v.y + v.z * v.z + v.w * v.w;
  }
  __shared__ float ls[256], lq[256];
  ls[threadIdx.x] = s;
  lq[threadIdx.x] = s2;
  __syncthreads();
  for (int st = 128; st > 0; st >>= 1) {
    if ((int)threadIdx.x < st) {
      ls[threadIdx.x] += ls[threadIdx.x + st];
      lq[threadIdx.x] += lq[threadIdx.x + st];
    }
    __syncthreads();
  }
  if (threadIdx.x == 0) {
    const float inv = 1.0f / (float)(CPG * N_TOK);
    float mu  = ls[0] * inv;
    float var = lq[0] * inv - mu * mu;
    stats[2 * g]     = mu;
    stats[2 * g + 1] = rsqrtf(var + GN_EPS);
  }
}

// ---- K2: normalize + transpose-pack to token-major bf16 -------------------

__global__ void norm_pack_kernel(const float* __restrict__ x,
                                 const float* __restrict__ gamma,
                                 const float* __restrict__ beta,
                                 const float* __restrict__ stats,
                                 bf16_t* __restrict__ hn_tm) {
  __shared__ float tile[32][33];
  const int tok0 = blockIdx.x * 32;
  const int ch0  = blockIdx.y * 32;
  const int tx = threadIdx.x, ty = threadIdx.y;
#pragma unroll
  for (int i = 0; i < 4; ++i) {
    int chl = ty + i * 8;
    int ch  = ch0 + chl;
    int g   = ch >> 4;
    float mu = stats[2 * g], rstd = stats[2 * g + 1];
    float sc = rstd * gamma[ch];
    float sh = beta[ch] - mu * sc;
    float v  = x[(unsigned)ch * N_TOK + tok0 + tx];
    tile[chl][tx] = v * sc + sh;
  }
  __syncthreads();
#pragma unroll
  for (int i = 0; i < 4; ++i) {
    int row = ty + i * 8;
    hn_tm[(unsigned)(tok0 + row) * C_CH + ch0 + tx] = (bf16_t)tile[tx][row];
  }
}

// ---- TDM-staged GEMM core (shared by both orientations) -------------------
// Per K-step: TDM loads the 128x32 A tile and 128x32 B tile into LDS
// (double-buffered, pad -> 80B pitch), waves compute 16 WMMAs each.

#define GEMM_PROLOGUE()                                                        \
  __shared__ bf16_t At[2][128 * LDS_PITCH];                                    \
  __shared__ bf16_t Bt[2][128 * LDS_PITCH];                                    \
  const int lane = threadIdx.x & 31;                                           \
  const int warp = threadIdx.x >> 5;                                           \
  const int ln   = lane & 15;                                                  \
  const int half = lane >> 4;                                                  \
  const unsigned mblk = blockIdx.x * 128;                                      \
  const unsigned nblk = blockIdx.y * 128;                                      \
  const unsigned wm = (warp >> 1) * 64; /* wave row offset in A tile */        \
  const unsigned wn = (warp & 1) * 64;  /* wave row offset in B tile */        \
  v8f acc[4][4];                                                               \
  const v8f vzero = {0.f, 0.f, 0.f, 0.f, 0.f, 0.f, 0.f, 0.f};                  \
  _Pragma("unroll") for (int i = 0; i < 4; ++i)                                \
      _Pragma("unroll") for (int j = 0; j < 4; ++j) acc[i][j] = vzero;

#define GEMM_MAINLOOP(Asrc, Arows, Bsrc, Brows)                                \
  if (warp == 0) {                                                             \
    tdm_load_2d((unsigned)(size_t)&At[0][0], (Asrc) + (size_t)mblk * C_CH,     \
                C_CH, (Arows), 32, 128, C_CH);                                 \
    tdm_load_2d((unsigned)(size_t)&Bt[0][0], (Bsrc) + (size_t)nblk * C_CH,     \
                C_CH, (Brows), 32, 128, C_CH);                                 \
  }                                                                            \
  for (int kk = 0; kk < 16; ++kk) {                                            \
    const int cur = kk & 1;                                                    \
    if (warp == 0) __builtin_amdgcn_s_wait_tensorcnt(0);                       \
    __syncthreads(); /* tile kk visible to all waves */                        \
    if (kk + 1 < 16 && warp == 0) {                                            \
      const unsigned k1 = (unsigned)(kk + 1) * 32u;                            \
      tdm_load_2d((unsigned)(size_t)&At[cur ^ 1][0],                           \
                  (Asrc) + (size_t)mblk * C_CH + k1, C_CH, (Arows), 32, 128,   \
                  C_CH);                                                       \
      tdm_load_2d((unsigned)(size_t)&Bt[cur ^ 1][0],                           \
                  (Bsrc) + (size_t)nblk * C_CH + k1, C_CH, (Brows), 32, 128,   \
                  C_CH);                                                       \
    }                                                                          \
    v16bf a[4], b[4];                                                          \
    _Pragma("unroll") for (int i = 0; i < 4; ++i)                              \
        a[i] = load_a_frag_lds(&At[cur][(wm + i * 16 + ln) * LDS_PITCH], lane);\
    _Pragma("unroll") for (int j = 0; j < 4; ++j)                              \
        b[j] = load_b_frag_lds(&Bt[cur][(wn + j * 16 + ln) * LDS_PITCH], lane);\
    _Pragma("unroll") for (int i = 0; i < 4; ++i)                              \
        _Pragma("unroll") for (int j = 0; j < 4; ++j)                          \
            acc[i][j] = wmma_bf16(a[i], b[j], acc[i][j]);                      \
    __syncthreads(); /* buffer cur may be refilled at kk+2 */                  \
  }

// ---- K3a: GEMM, token-major output (Q and K) ------------------------------
// out_tm[tok, oc] = (sum_ic hn_tm[tok, ic] * W[oc, ic] + bias[oc]) * scale

__global__ __launch_bounds__(128)
__attribute__((amdgpu_waves_per_eu(1, 4)))
void gemm_tm_kernel(const bf16_t* __restrict__ A,
                    const bf16_t* __restrict__ W,
                    const float* __restrict__ bias,
                    bf16_t* __restrict__ out,
                    float scale) {
  GEMM_PROLOGUE()
  GEMM_MAINLOOP(A, N_TOK, W, C_CH)
  const unsigned m0 = mblk + wm, n0 = nblk + wn;
#pragma unroll
  for (int j = 0; j < 4; ++j) {
    float bn = bias[n0 + j * 16 + ln];
#pragma unroll
    for (int i = 0; i < 4; ++i)
#pragma unroll
      for (int r = 0; r < 8; ++r) {
        unsigned m = m0 + i * 16 + r + 8 * half;
        out[m * C_CH + n0 + j * 16 + ln] = (bf16_t)((acc[i][j][r] + bn) * scale);
      }
  }
}

// ---- K3b: GEMM, channel-major output (V and final projection) -------------
// acc[oc, tok] = sum_ic W[oc, ic] * Btm[tok, ic]
// mode 0: out_cm = bf16(acc + bias[oc]);  mode 1: out_f32 = acc + bias + resid

__global__ __launch_bounds__(128)
__attribute__((amdgpu_waves_per_eu(1, 4)))
void gemm_cm_kernel(const bf16_t* __restrict__ W,
                    const bf16_t* __restrict__ Btm,
                    const float* __restrict__ bias,
                    bf16_t* __restrict__ out_cm,
                    const float* __restrict__ resid,
                    float* __restrict__ out_f32,
                    int mode) {
  GEMM_PROLOGUE()
  GEMM_MAINLOOP(W, C_CH, Btm, N_TOK)
  const unsigned m0 = mblk + wm, n0 = nblk + wn;
#pragma unroll
  for (int i = 0; i < 4; ++i)
#pragma unroll
    for (int r = 0; r < 8; ++r) {
      unsigned m = m0 + i * 16 + r + 8 * half;
      float bm = bias[m];
#pragma unroll
      for (int j = 0; j < 4; ++j) {
        unsigned idx = m * N_TOK + n0 + j * 16 + ln;
        float v = acc[i][j][r] + bm;
        if (mode)
          out_f32[idx] = v + resid[idx];
        else
          out_cm[idx] = (bf16_t)v;
      }
    }
}

// ---- K4: per-frame flash attention, wave-pair split-K ---------------------

__global__ __launch_bounds__(128)
__attribute__((amdgpu_waves_per_eu(1, 4)))
void attn_kernel(const bf16_t* __restrict__ q_tm,
                 const bf16_t* __restrict__ k_tm,
                 const bf16_t* __restrict__ v_cm,
                 bf16_t* __restrict__ o_tm) {
  __shared__ float  Sx[4][1024];        // partial-S exchange (16KB)
  __shared__ bf16_t Pl[4][16 * 64];     // per-wave P tile copies (8KB)
  const int lane = threadIdx.x & 31;
  const int warp = threadIdx.x >> 5;
  const int pair = warp >> 1;
  const int hsel = warp & 1;
  const int ln   = lane & 15;
  const int half = lane >> 4;
  const int frame = blockIdx.y;
  const int q0 = frame * HW + blockIdx.x * 32 + pair * 16;
  const int kf = frame * HW;
  const unsigned chbase = (unsigned)hsel * 256u;
  const unsigned kcbase = (unsigned)hsel * 256u;

  v8f O[16];
  const v8f vzero = {0.f, 0.f, 0.f, 0.f, 0.f, 0.f, 0.f, 0.f};
#pragma unroll
  for (int s = 0; s < 16; ++s) O[s] = vzero;
  float mrow[8], lrow[8];
#pragma unroll
  for (int r = 0; r < 8; ++r) { mrow[r] = -1e30f; lrow[r] = 0.f; }

  const unsigned qrow = (unsigned)(q0 + ln) * C_CH + kcbase;

  for (int kb = 0; kb < HW; kb += 64) {
    const unsigned kt0 = (unsigned)(kf + kb);

    v8f S[4];
#pragma unroll
    for (int j = 0; j < 4; ++j) S[j] = vzero;
#pragma unroll
    for (unsigned kc = 0; kc < 256; kc += 32) {
      v16bf aq = load_a_frag(q_tm, qrow + kc, lane);
#pragma unroll
      for (int j = 0; j < 4; ++j) {
        v16bf bk = load_b_frag(k_tm, (kt0 + j * 16 + ln) * C_CH + kcbase + kc, lane);
        S[j] = wmma_bf16(aq, bk, S[j]);
      }
    }

#pragma unroll
    for (int j = 0; j < 4; ++j)
#pragma unroll
      for (int r = 0; r < 8; ++r)
        Sx[warp][(j * 8 + r) * 32 + lane] = S[j][r];
    __syncthreads();
#pragma unroll
    for (int j = 0; j < 4; ++j)
#pragma unroll
      for (int r = 0; r < 8; ++r)
        S[j][r] += Sx[warp ^ 1][(j * 8 + r) * 32 + lane];
    __syncthreads();

    float alpha[8];
#pragma unroll
    for (int r = 0; r < 8; ++r) {
      float t = fmaxf(fmaxf(S[0][r], S[1][r]), fmaxf(S[2][r], S[3][r]));
#pragma unroll
      for (int msk = 1; msk < 16; msk <<= 1) t = fmaxf(t, __shfl_xor(t, msk, 32));
      float mn = fmaxf(mrow[r], t);
      alpha[r] = __expf(mrow[r] - mn);
      mrow[r] = mn;
    }
    float rs[8];
#pragma unroll
    for (int r = 0; r < 8; ++r) rs[r] = 0.f;
#pragma unroll
    for (int j = 0; j < 4; ++j)
#pragma unroll
      for (int r = 0; r < 8; ++r) {
        float p = __expf(S[j][r] - mrow[r]);
        rs[r] += p;
        Pl[warp][(r + 8 * half) * 64 + j * 16 + ln] = (bf16_t)p;
      }
#pragma unroll
    for (int r = 0; r < 8; ++r) {
      float t = rs[r];
#pragma unroll
      for (int msk = 1; msk < 16; msk <<= 1) t += __shfl_xor(t, msk, 32);
      lrow[r] = lrow[r] * alpha[r] + t;
    }

#pragma unroll
    for (int s = 0; s < 16; ++s)
#pragma unroll
      for (int r = 0; r < 8; ++r) O[s][r] *= alpha[r];

#pragma unroll
    for (int kc2 = 0; kc2 < 2; ++kc2) {
      v16bf ap = load_a_frag_lds(&Pl[warp][ln * 64 + kc2 * 32], lane);
#pragma unroll
      for (int s = 0; s < 16; ++s) {
        v16bf bv = load_b_frag(
            v_cm, (chbase + s * 16 + ln) * (unsigned)N_TOK + kt0 + kc2 * 32, lane);
        O[s] = wmma_bf16(ap, bv, O[s]);
      }
    }
  }

#pragma unroll
  for (int r = 0; r < 8; ++r) lrow[r] = 1.0f / lrow[r];
#pragma unroll
  for (int s = 0; s < 16; ++s)
#pragma unroll
    for (int r = 0; r < 8; ++r) {
      unsigned tok = (unsigned)(q0 + r + 8 * half);
      o_tm[tok * C_CH + chbase + s * 16 + ln] = (bf16_t)(O[s][r] * lrow[r]);
    }
}

// ---------------------------------------------------------------------------

extern "C" void kernel_launch(void* const* d_in, const int* in_sizes, int n_in,
                              void* d_out, int out_size, void* d_ws, size_t ws_size,
                              hipStream_t stream) {
  const float* x     = (const float*)d_in[0];
  const float* gamma = (const float*)d_in[1];
  const float* beta  = (const float*)d_in[2];
  const float* wq    = (const float*)d_in[3];
  const float* bq    = (const float*)d_in[4];
  const float* wk    = (const float*)d_in[5];
  const float* bk    = (const float*)d_in[6];
  const float* wv    = (const float*)d_in[7];
  const float* bv    = (const float*)d_in[8];
  const float* wo    = (const float*)d_in[9];
  const float* bo    = (const float*)d_in[10];
  float* out = (float*)d_out;

  char* ws = (char*)d_ws;
  size_t off = 0;
  float* stats = (float*)(ws + off); off += 1024;
  bf16_t* wq_bf = (bf16_t*)(ws + off); off += (size_t)C_CH * C_CH * 2;
  bf16_t* wk_bf = (bf16_t*)(ws + off); off += (size_t)C_CH * C_CH * 2;
  bf16_t* wv_bf = (bf16_t*)(ws + off); off += (size_t)C_CH * C_CH * 2;
  bf16_t* wo_bf = (bf16_t*)(ws + off); off += (size_t)C_CH * C_CH * 2;
  off = (off + 4095) & ~(size_t)4095;
  bf16_t* hn  = (bf16_t*)(ws + off); off += (size_t)N_TOK * C_CH * 2;  // [N,C]
  bf16_t* qb  = (bf16_t*)(ws + off); off += (size_t)N_TOK * C_CH * 2;  // [N,C], reused as O
  bf16_t* kb2 = (bf16_t*)(ws + off); off += (size_t)N_TOK * C_CH * 2;  // [N,C]
  bf16_t* vb  = (bf16_t*)(ws + off); off += (size_t)N_TOK * C_CH * 2;  // [C,N]

  const int wn = C_CH * C_CH;
  cvt_bf16_kernel<<<(wn + 255) / 256, 256, 0, stream>>>(wq, wq_bf, wn);
  cvt_bf16_kernel<<<(wn + 255) / 256, 256, 0, stream>>>(wk, wk_bf, wn);
  cvt_bf16_kernel<<<(wn + 255) / 256, 256, 0, stream>>>(wv, wv_bf, wn);
  cvt_bf16_kernel<<<(wn + 255) / 256, 256, 0, stream>>>(wo, wo_bf, wn);

  gn_stats_kernel<<<NUM_GROUPS, 256, 0, stream>>>(x, stats);
  norm_pack_kernel<<<dim3(N_TOK / 32, C_CH / 32), dim3(32, 8), 0, stream>>>(
      x, gamma, beta, stats, hn);

  const float scale = 0.044194173824159216f;   // 512^-0.5, folded into Q
  gemm_tm_kernel<<<dim3(N_TOK / 128, C_CH / 128), 128, 0, stream>>>(
      hn, wq_bf, bq, qb, scale);
  gemm_tm_kernel<<<dim3(N_TOK / 128, C_CH / 128), 128, 0, stream>>>(
      hn, wk_bf, bk, kb2, 1.0f);
  gemm_cm_kernel<<<dim3(C_CH / 128, N_TOK / 128), 128, 0, stream>>>(
      wv_bf, hn, bv, vb, nullptr, nullptr, 0);

  attn_kernel<<<dim3(HW / 32, T_FRAMES), 128, 0, stream>>>(qb, kb2, vb, qb);

  gemm_cm_kernel<<<dim3(C_CH / 128, N_TOK / 128), 128, 0, stream>>>(
      wo_bf, qb, bo, nullptr, x, out, 1);
}
